// PrunableQwenImageTransformerBlock_86586540687518
// MI455X (gfx1250) — matrix operand order repaired and checked by reading, not compiled, not run on hardware
//
#include <hip/hip_runtime.h>

#define D_MODEL 3072
#define NHEADS  24
#define HEADDIM 128
#define L_DEN   1024
#define L_IMG   3072
#define S_TXT   256
#define LQ      (S_TXT + L_DEN)          /* 1280 */
#define LK      (S_TXT + L_DEN + L_IMG)  /* 4352 */

typedef __attribute__((ext_vector_type(16))) __bf16 v16bf;
typedef __attribute__((ext_vector_type(8)))  __bf16 v8bf;
typedef __attribute__((ext_vector_type(8)))  float  v8f;
typedef __attribute__((ext_vector_type(4)))  int    v4i;

union AFrag { v16bf v; v8bf h[2]; };

__device__ __forceinline__ v8f zero8() {
    v8f z = {0.f,0.f,0.f,0.f,0.f,0.f,0.f,0.f};
    return z;
}

// ---------------------------------------------------------------------------
// gfx1250 async global->LDS copy (ASYNCcnt path), guarded so the source
// still compiles on toolchains without the builtins.  The builtin takes
// (global v4i*, local v4i*, imm offset, imm cpol).
// ---------------------------------------------------------------------------
#if __has_builtin(__builtin_amdgcn_global_load_async_to_lds_b128) && \
    __has_builtin(__builtin_amdgcn_s_wait_asynccnt)
#define HAS_ASYNC_LDS 1
typedef __attribute__((address_space(1))) v4i g_v4i;
typedef __attribute__((address_space(3))) v4i l_v4i;
__device__ __forceinline__ void async_cp16(const void* g, void* l) {
    __builtin_amdgcn_global_load_async_to_lds_b128((g_v4i*)g, (l_v4i*)l, 0, 0);
}
__device__ __forceinline__ void async_wait0() {
    __builtin_amdgcn_s_wait_asynccnt(0);
}
#else
#define HAS_ASYNC_LDS 0
#endif

// ---------------------------------------------------------------------------
// fp32 -> bf16 elementwise convert
// ---------------------------------------------------------------------------
__global__ void cvt_f32_bf16(const float* __restrict__ src,
                             __bf16* __restrict__ dst, int n) {
    for (int i = blockIdx.x * blockDim.x + threadIdx.x; i < n;
         i += gridDim.x * blockDim.x)
        dst[i] = (__bf16)src[i];
}

// ---------------------------------------------------------------------------
// cached image K: per-head RMS + RoPE (positions 1024..4095), fp32 -> bf16
// grid = (L_IMG, NHEADS), block = 32 (one wave reduces one 128-dim head row)
// ---------------------------------------------------------------------------
__global__ __launch_bounds__(32) void rms_rope_cached_k(
    const float* __restrict__ src, const float* __restrict__ rmsw,
    const float* __restrict__ fcos, const float* __restrict__ fsin,
    __bf16* __restrict__ dst) {
    const int row = blockIdx.x;
    const int h   = blockIdx.y;
    const int t   = threadIdx.x;
    const int d0  = t * 4;
    const float* p = src + (size_t)row * D_MODEL + h * HEADDIM + d0;
    float v0 = p[0], v1 = p[1], v2 = p[2], v3 = p[3];
    float ss = v0*v0 + v1*v1 + v2*v2 + v3*v3;
    ss += __shfl_xor(ss, 1, 32);  ss += __shfl_xor(ss, 2, 32);
    ss += __shfl_xor(ss, 4, 32);  ss += __shfl_xor(ss, 8, 32);
    ss += __shfl_xor(ss, 16, 32);
    const float sc = rsqrtf(ss * (1.f/128.f) + 1e-6f);
    v0 *= sc * rmsw[d0];   v1 *= sc * rmsw[d0+1];
    v2 *= sc * rmsw[d0+2]; v3 *= sc * rmsw[d0+3];
    const int pos = L_DEN + row;
    const int f0 = d0 >> 1;
    const float c0 = fcos[pos*64 + f0],     s0 = fsin[pos*64 + f0];
    const float c1 = fcos[pos*64 + f0 + 1], s1 = fsin[pos*64 + f0 + 1];
    const float o0 = v0*c0 - v1*s0, o1 = v0*s0 + v1*c0;
    const float o2 = v2*c1 - v3*s1, o3 = v2*s1 + v3*c1;
    __bf16* q = dst + (size_t)(S_TXT + L_DEN + row) * D_MODEL + h * HEADDIM + d0;
    q[0] = (__bf16)o0; q[1] = (__bf16)o1; q[2] = (__bf16)o2; q[3] = (__bf16)o3;
}

// ---------------------------------------------------------------------------
// Fused GEMM: dst = X(bf16) @ W(fp32->bf16) + bias, optional per-head RMS +
// RoPE epilogue.  Block = 128 thr (4 waves), tile = 64 rows x 128 cols.
// A tile async-staged global->LDS; W tile through VGPRs (batched loads,
// fp32->bf16 cvt) into transposed LDS; 8 B fragments batched ahead of an
// 8-deep WMMA burst.
// ---------------------------------------------------------------------------
__global__ __launch_bounds__(128) void gemm_fused(
    const __bf16* __restrict__ X, const float* __restrict__ W,
    const float* __restrict__ bias, const float* __restrict__ rmsw,
    const float* __restrict__ fcos, const float* __restrict__ fsin,
    int pos_off, __bf16* __restrict__ dst_bf, float* __restrict__ dst_f,
    int dst_row_off) {
    __shared__ __bf16 Bs[128 * 40];   // W tile, transposed [n][k], padded
#if HAS_ASYNC_LDS
    __shared__ __bf16 As[64 * 32];    // X tile, row-major [m][k]
#endif

    const int tid  = threadIdx.x;
    const int wave = tid >> 5;
    const int lane = tid & 31;
    const int lrow = lane & 15;
    const int hi8  = (lane >> 4) & 1;
    const int mb   = blockIdx.x * 64;
    const int m0   = mb + wave * 16;
    const int n0   = blockIdx.y * 128;

    v8f acc[8];
    #pragma unroll
    for (int t = 0; t < 8; t++) acc[t] = zero8();

    const __bf16* xrow = X + (size_t)(m0 + lrow) * D_MODEL;
    (void)xrow;

    for (int k0 = 0; k0 < D_MODEL; k0 += 32) {
#if HAS_ASYNC_LDS
        // async stage 64x32 bf16 X tile, row-major (256 x 16B chunks)
        {
            #pragma unroll
            for (int j = 0; j < 2; j++) {
                const int idx = tid + j * 128;       // 0..255
                const int r   = idx >> 2;            // row 0..63
                const int s4  = (idx & 3) * 8;       // 0,8,16,24 elems
                async_cp16(X + (size_t)(mb + r) * D_MODEL + k0 + s4,
                           &As[r * 32 + s4]);
            }
        }
#endif
        // stage 32x128 W tile into LDS as bf16, transposed for frag reads.
        // Batch all global loads first so they pipeline behind one wait.
        {
            float4 w4[8];
            #pragma unroll
            for (int j = 0; j < 8; j++) {
                const int idx = tid + j * 128;       // 0..1023
                const int k   = idx >> 5;            // 0..31
                const int n4  = (idx & 31) * 4;      // 0..124
                w4[j] = *(const float4*)&W[(size_t)(k0 + k) * D_MODEL + n0 + n4];
            }
            #pragma unroll
            for (int j = 0; j < 8; j++) {
                const int idx = tid + j * 128;
                const int k   = idx >> 5;
                const int n4  = (idx & 31) * 4;
                Bs[(n4 + 0) * 40 + k] = (__bf16)w4[j].x;
                Bs[(n4 + 1) * 40 + k] = (__bf16)w4[j].y;
                Bs[(n4 + 2) * 40 + k] = (__bf16)w4[j].z;
                Bs[(n4 + 3) * 40 + k] = (__bf16)w4[j].w;
            }
            if (k0 + 32 < D_MODEL)
                __builtin_prefetch(&W[(size_t)(k0 + 32) * D_MODEL + n0 + tid], 0, 0);
        }
#if HAS_ASYNC_LDS
        async_wait0();
#endif
        __syncthreads();

        AFrag a;   // 16x32 bf16 A fragment (ISA 16-bit A layout)
#if HAS_ASYNC_LDS
        {
            const __bf16* ap = &As[(wave * 16 + lrow) * 32];
            a.h[0] = *(const v8bf*)(ap + hi8 * 8);
            a.h[1] = *(const v8bf*)(ap + 16 + hi8 * 8);
        }
#else
        a.h[0] = *(const v8bf*)(xrow + k0 + hi8 * 8);
        a.h[1] = *(const v8bf*)(xrow + k0 + 16 + hi8 * 8);
#endif

        // batch all 8 B fragments, then an 8-deep WMMA burst
        AFrag b[8];
        #pragma unroll
        for (int t = 0; t < 8; t++) {
            const __bf16* bp = &Bs[(t * 16 + lrow) * 40 + hi8 * 8];
            b[t].h[0] = *(const v8bf*)(bp);
            b[t].h[1] = *(const v8bf*)(bp + 16);
        }
        #pragma unroll
        for (int t = 0; t < 8; t++)
            acc[t] = __builtin_amdgcn_wmma_f32_16x16x32_bf16(
                false, a.v, false, b[t].v, (short)0, acc[t], false, false);
        __syncthreads();
    }

    // ---- epilogue: bias ----
    #pragma unroll
    for (int t = 0; t < 8; t++) {
        const float bv = bias[n0 + t * 16 + lrow];
        #pragma unroll
        for (int r = 0; r < 8; r++) acc[t][r] += bv;
    }

    // ---- optional per-head RMS (row reduce across 16 lanes + 8 tiles) ----
    float rscale[8];
    #pragma unroll
    for (int r = 0; r < 8; r++) rscale[r] = 1.f;
    if (rmsw) {
        #pragma unroll
        for (int r = 0; r < 8; r++) {
            float ss = 0.f;
            #pragma unroll
            for (int t = 0; t < 8; t++) { float v = acc[t][r]; ss += v * v; }
            ss += __shfl_xor(ss, 1, 32); ss += __shfl_xor(ss, 2, 32);
            ss += __shfl_xor(ss, 4, 32); ss += __shfl_xor(ss, 8, 32);
            rscale[r] = rsqrtf(ss * (1.f/128.f) + 1e-6f);
        }
    }

    const bool even = ((lane & 1) == 0);
    #pragma unroll
    for (int t = 0; t < 8; t++) {
        const int d = t * 16 + lrow;                // 0..127 inside head
        const float w = rmsw ? rmsw[d] : 1.f;
        #pragma unroll
        for (int r = 0; r < 8; r++) {
            float v = acc[t][r];
            if (rmsw) v *= rscale[r] * w;
            if (fcos) {   // RoPE: (even,odd) column pair lives in lane L, L^1
                const int pos = m0 + r + hi8 * 8 + pos_off;
                const int f = d >> 1;
                const float c = fcos[pos * 64 + f];
                const float s = fsin[pos * 64 + f];
                const float p = __shfl_xor(v, 1, 32);
                v = even ? (v * c - p * s) : (p * s + v * c);
            }
            const int grow = dst_row_off + m0 + r + hi8 * 8;
            const int gcol = n0 + d;
            if (dst_bf) dst_bf[(size_t)grow * D_MODEL + gcol] = (__bf16)v;
            else        dst_f [(size_t)grow * D_MODEL + gcol] = v;
        }
    }
}

// ---------------------------------------------------------------------------
// Flash attention: one wave per (16-query tile, head).  Online softmax over
// 32-key blocks; QK^T and PV both via v_wmma_f32_16x16x32_bf16.
// K tile async-staged global->LDS (row-major matches the B-fragment read
// order); V tile transposed through VGPRs for the PV operand.  Fragment
// loads batched ahead of WMMA bursts.
// ---------------------------------------------------------------------------
__global__ __launch_bounds__(32) void flash_attn(
    const __bf16* __restrict__ Q, const __bf16* __restrict__ K,
    const __bf16* __restrict__ V, const int* __restrict__ mask,
    __bf16* __restrict__ O) {
    __shared__ __bf16 Ps[16 * 32];    // P tile, C-layout -> A-layout transpose
    __shared__ __bf16 Vs[128 * 40];   // V tile transposed [d][k], padded
#if HAS_ASYNC_LDS
    __shared__ __bf16 Ks[32 * 128];   // K tile, row-major [key][c]
#endif

    const int lane   = threadIdx.x;
    const int lrow   = lane & 15;
    const int hi8    = (lane >> 4) & 1;
    const int koff16 = hi8 * 16;
    const int q0     = blockIdx.x * 16;
    const int hbase  = blockIdx.y * HEADDIM;
    const float SC   = 0.08838834764831845f;   // 1/sqrt(128)

    AFrag aq[4];   // Q tile 16x128 as 4 A-fragments, loaded once
    {
        const __bf16* qrow = Q + (size_t)(q0 + lrow) * D_MODEL + hbase;
        #pragma unroll
        for (int kk = 0; kk < 4; kk++) {
            aq[kk].h[0] = *(const v8bf*)(qrow + kk * 32 + hi8 * 8);
            aq[kk].h[1] = *(const v8bf*)(qrow + kk * 32 + 16 + hi8 * 8);
        }
    }

    float m_s[8], l_s[8];
    v8f o[8];
    #pragma unroll
    for (int r = 0; r < 8; r++) { m_s[r] = -3.0e38f; l_s[r] = 0.f; }
    #pragma unroll
    for (int t = 0; t < 8; t++) o[t] = zero8();

    for (int kb = 0; kb < LK; kb += 32) {
#if HAS_ASYNC_LDS
        // async stage 32x128 bf16 K tile, row-major (512 x 16B chunks)
        {
            #pragma unroll
            for (int j = 0; j < 16; j++) {
                const int idx = lane + j * 32;       // 0..511
                const int r   = idx >> 4;            // key row 0..31
                const int s8  = (idx & 15) * 8;      // 0..120 elems
                async_cp16(K + (size_t)(kb + r) * D_MODEL + hbase + s8,
                           &Ks[r * 128 + s8]);
            }
        }
#endif
        // stage 32x128 V tile into LDS transposed
        {
            const __bf16* vrow = V + (size_t)(kb + lane) * D_MODEL + hbase;
            #pragma unroll
            for (int dc = 0; dc < 16; dc++) {
                v8bf tmp = *(const v8bf*)(vrow + dc * 8);
                #pragma unroll
                for (int e = 0; e < 8; e++)
                    Vs[(dc * 8 + e) * 40 + lane] = tmp[e];
            }
        }
#if HAS_ASYNC_LDS
        async_wait0();
#endif

        // S = Q @ K^T for two 16-key subtiles: batch 8 frags, then 8 WMMAs
        v8f s0 = zero8(), s1 = zero8();
        AFrag bk0[4], bk1[4];
        #pragma unroll
        for (int kk = 0; kk < 4; kk++) {
#if HAS_ASYNC_LDS
            const __bf16* k0p = &Ks[lrow * 128 + kk * 32 + hi8 * 8];
            const __bf16* k1p = &Ks[(16 + lrow) * 128 + kk * 32 + hi8 * 8];
#else
            const __bf16* k0p = K + (size_t)(kb + lrow) * D_MODEL + hbase + kk * 32 + hi8 * 8;
            const __bf16* k1p = K + (size_t)(kb + 16 + lrow) * D_MODEL + hbase + kk * 32 + hi8 * 8;
#endif
            bk0[kk].h[0] = *(const v8bf*)(k0p); bk0[kk].h[1] = *(const v8bf*)(k0p + 16);
            bk1[kk].h[0] = *(const v8bf*)(k1p); bk1[kk].h[1] = *(const v8bf*)(k1p + 16);
        }
        #pragma unroll
        for (int kk = 0; kk < 4; kk++) {
            s0 = __builtin_amdgcn_wmma_f32_16x16x32_bf16(
                false, aq[kk].v, false, bk0[kk].v, (short)0, s0, false, false);
            s1 = __builtin_amdgcn_wmma_f32_16x16x32_bf16(
                false, aq[kk].v, false, bk1[kk].v, (short)0, s1, false, false);
        }

        const int key0 = kb + lrow, key1 = kb + 16 + lrow;
        const float bias0 = (key0 < S_TXT) ? (mask[key0] ? 0.f : -1e30f) : 0.f;
        const float bias1 = (key1 < S_TXT) ? (mask[key1] ? 0.f : -1e30f) : 0.f;

        // online softmax per output row (8 rows per half-wave)
        #pragma unroll
        for (int r = 0; r < 8; r++) {
            float a = s0[r] * SC + bias0;
            float b = s1[r] * SC + bias1;
            float mx = fmaxf(a, b);
            mx = fmaxf(mx, __shfl_xor(mx, 1, 32));
            mx = fmaxf(mx, __shfl_xor(mx, 2, 32));
            mx = fmaxf(mx, __shfl_xor(mx, 4, 32));
            mx = fmaxf(mx, __shfl_xor(mx, 8, 32));
            const float nm = fmaxf(m_s[r], mx);
            const float p0 = __expf(a - nm), p1 = __expf(b - nm);
            float rs = p0 + p1;
            rs += __shfl_xor(rs, 1, 32); rs += __shfl_xor(rs, 2, 32);
            rs += __shfl_xor(rs, 4, 32); rs += __shfl_xor(rs, 8, 32);
            const float al = __expf(m_s[r] - nm);
            l_s[r] = l_s[r] * al + rs;
            m_s[r] = nm;
            #pragma unroll
            for (int t = 0; t < 8; t++) o[t][r] *= al;
            const int prow = r + hi8 * 8;
            Ps[prow * 32 + lrow]      = (__bf16)p0;
            Ps[prow * 32 + 16 + lrow] = (__bf16)p1;
        }
        __syncthreads();

        // O += P @ V : batch 8 V fragments, then 8 WMMAs
        AFrag ap;
        ap.h[0] = *(const v8bf*)&Ps[lrow * 32 + hi8 * 8];
        ap.h[1] = *(const v8bf*)&Ps[lrow * 32 + 16 + hi8 * 8];
        AFrag bv[8];
        #pragma unroll
        for (int t = 0; t < 8; t++) {
            const __bf16* vp = &Vs[(t * 16 + lrow) * 40 + hi8 * 8];
            bv[t].h[0] = *(const v8bf*)(vp); bv[t].h[1] = *(const v8bf*)(vp + 16);
        }
        #pragma unroll
        for (int t = 0; t < 8; t++)
            o[t] = __builtin_amdgcn_wmma_f32_16x16x32_bf16(
                false, ap.v, false, bv[t].v, (short)0, o[t], false, false);
        __syncthreads();
    }

    #pragma unroll
    for (int t = 0; t < 8; t++) {
        #pragma unroll
        for (int r = 0; r < 8; r++) {
            const float val = o[t][r] / l_s[r];
            O[(size_t)(q0 + r + hi8 * 8) * D_MODEL + hbase + t * 16 + lrow] =
                (__bf16)val;
        }
    }
}

// ---------------------------------------------------------------------------
extern "C" void kernel_launch(void* const* d_in, const int* in_sizes, int n_in,
                              void* d_out, int out_size, void* d_ws,
                              size_t ws_size, hipStream_t stream) {
    (void)in_sizes; (void)n_in; (void)out_size; (void)ws_size;
    const float* hidden = (const float*)d_in[0];
    const float* enc    = (const float*)d_in[1];
    const int*   msk    = (const int*)d_in[2];
    const float* cK     = (const float*)d_in[3];
    const float* cV     = (const float*)d_in[4];
    const float* icos   = (const float*)d_in[5];
    const float* isin   = (const float*)d_in[6];
    const float* tcos   = (const float*)d_in[7];
    const float* tsin   = (const float*)d_in[8];
    const float* Wq     = (const float*)d_in[9];
    const float* bq     = (const float*)d_in[10];
    const float* Wk     = (const float*)d_in[11];
    const float* bk     = (const float*)d_in[12];
    const float* Wv     = (const float*)d_in[13];
    const float* bv     = (const float*)d_in[14];
    const float* Wqa    = (const float*)d_in[15];
    const float* bqa    = (const float*)d_in[16];
    const float* Wka    = (const float*)d_in[17];
    const float* bka    = (const float*)d_in[18];
    const float* Wva    = (const float*)d_in[19];
    const float* bva    = (const float*)d_in[20];
    const float* nq     = (const float*)d_in[21];
    const float* nk     = (const float*)d_in[22];
    const float* naq    = (const float*)d_in[23];
    const float* nak    = (const float*)d_in[24];
    const float* Wo     = (const float*)d_in[25];
    const float* bo     = (const float*)d_in[26];
    const float* Woa    = (const float*)d_in[27];
    const float* boa    = (const float*)d_in[28];
    float* out = (float*)d_out;

    size_t off = 0;
    auto alloc = [&](size_t elems) -> __bf16* {
        __bf16* p = (__bf16*)((char*)d_ws + off);
        off += ((elems * sizeof(__bf16)) + 255) & ~(size_t)255;
        return p;
    };
    __bf16* Xd = alloc((size_t)L_DEN * D_MODEL);
    __bf16* Xe = alloc((size_t)S_TXT * D_MODEL);
    __bf16* Qb = alloc((size_t)LQ * D_MODEL);
    __bf16* Kb = alloc((size_t)LK * D_MODEL);
    __bf16* Vb = alloc((size_t)LK * D_MODEL);
    __bf16* Ob = alloc((size_t)LQ * D_MODEL);

    // activations / cached tensors -> bf16 (cached K also gets RMS+RoPE)
    cvt_f32_bf16<<<4096, 256, 0, stream>>>(hidden, Xd, L_DEN * D_MODEL);
    cvt_f32_bf16<<<1024, 256, 0, stream>>>(enc, Xe, S_TXT * D_MODEL);
    cvt_f32_bf16<<<8192, 256, 0, stream>>>(
        cV, Vb + (size_t)(S_TXT + L_DEN) * D_MODEL, L_IMG * D_MODEL);
    rms_rope_cached_k<<<dim3(L_IMG, NHEADS), 32, 0, stream>>>(
        cK, nk, icos, isin, Kb);

    const dim3 gD(L_DEN / 64, NHEADS), gT(S_TXT / 64, NHEADS);
    // image stream q,k,v (queries/keys at rows 256.., RoPE positions 0..1023)
    gemm_fused<<<gD, 128, 0, stream>>>(Xd, Wq, bq, nq, icos, isin, 0, Qb, nullptr, S_TXT);
    gemm_fused<<<gD, 128, 0, stream>>>(Xd, Wk, bk, nk, icos, isin, 0, Kb, nullptr, S_TXT);
    gemm_fused<<<gD, 128, 0, stream>>>(Xd, Wv, bv, nullptr, nullptr, nullptr, 0, Vb, nullptr, S_TXT);
    // text stream at rows 0..255
    gemm_fused<<<gT, 128, 0, stream>>>(Xe, Wqa, bqa, naq, tcos, tsin, 0, Qb, nullptr, 0);
    gemm_fused<<<gT, 128, 0, stream>>>(Xe, Wka, bka, nak, tcos, tsin, 0, Kb, nullptr, 0);
    gemm_fused<<<gT, 128, 0, stream>>>(Xe, Wva, bva, nullptr, nullptr, nullptr, 0, Vb, nullptr, 0);

    flash_attn<<<dim3(LQ / 16, NHEADS), 32, 0, stream>>>(Qb, Kb, Vb, msk, Ob);

    // output projections: img rows -> d_out[0..], txt rows -> d_out[1024*3072..]
    gemm_fused<<<gD, 128, 0, stream>>>(Ob + (size_t)S_TXT * D_MODEL, Wo, bo,
                                       nullptr, nullptr, nullptr, 0,
                                       nullptr, out, 0);
    gemm_fused<<<gT, 128, 0, stream>>>(Ob, Woa, boa,
                                       nullptr, nullptr, nullptr, 0,
                                       nullptr, out + (size_t)L_DEN * D_MODEL, 0);
}
